// MambaSSM_59253368816275
// MI455X (gfx1250) — compile-verified, matrix operand order checked by reading, not run on hardware
//
#include <hip/hip_runtime.h>
#include <hip/hip_bf16.h>
#include <stdint.h>

// Mamba-2 SSD forward for MI455X (gfx1250, wave32, WMMA).
// All matmuls run on v_wmma_f32_16x16x32_f16 (f16 operands, fp32 accumulate).
// Every exponential in the SSD is exp(non-positive) by construction, so f16
// operand ranges are safe. Weights in f16 (35MB) live in the 192MB L2.
// The projection GEMMs stage operand tiles into LDS with the gfx1250 async
// copy path (GLOBAL_LOAD_ASYNC_TO_LDS_B128 + s_wait_asynccnt) when the
// toolchain exposes the builtins, falling back to synchronous staging.

typedef __attribute__((ext_vector_type(16))) _Float16 v16h;
typedef __attribute__((ext_vector_type(8)))  _Float16 v8h;
typedef __attribute__((ext_vector_type(8)))  float    v8f;
typedef __attribute__((ext_vector_type(4)))  int      v4i;

#define DEV __device__ __forceinline__

#if defined(__has_builtin)
#if __has_builtin(__builtin_amdgcn_global_load_async_to_lds_b128) && \
    __has_builtin(__builtin_amdgcn_s_wait_asynccnt)
#define USE_ASYNC_LDS 1
#endif
#endif
#ifndef USE_ASYNC_LDS
#define USE_ASYNC_LDS 0
#endif

// Copy 16 bytes global -> LDS.  Async (ASYNCcnt-tracked, no VGPR round trip)
// when available.  Builtin signature (from hipcc diagnostics): param 0 is
// addrspace(1) int4* (prints as "__device__"), so cast accordingly.
DEV void cp16_to_lds(_Float16* l, const _Float16* g) {
#if USE_ASYNC_LDS
  __builtin_amdgcn_global_load_async_to_lds_b128(
      (__attribute__((address_space(1))) v4i*)g,
      (__attribute__((address_space(3))) v4i*)l, 0, 0);
#else
  *(v8h*)l = *(const v8h*)g;
#endif
}

DEV void wait_lds_cp() {
#if USE_ASYNC_LDS
  __builtin_amdgcn_s_wait_asynccnt(0);
#endif
}

DEV v8f wmma32(v16h a, v16h b, v8f c) {
  // (neg_a, A, neg_b, B, c_mod, C, reuse_a, reuse_b)
  return __builtin_amdgcn_wmma_f32_16x16x32_f16(false, a, false, b, (short)0, c,
                                                false, false);
}

// Load a 16x32 f16 WMMA A/B fragment from a row-major [rows, ld] matrix at
// (row0, k0).  Per CDNA5 ISA layout: lanes 0-15 hold rows M=lane with
// K = {k0+0..7, k0+16..23}; lanes 16-31 hold the same rows with K shifted +8.
// Two 16-byte loads per lane; ld*2, k0*2 are multiples of 16 at all call sites.
DEV v16h frag_load(const _Float16* p, int ld, int row0, int k0) {
  const int lane = threadIdx.x & 31;
  const _Float16* base =
      p + (size_t)(row0 + (lane & 15)) * ld + k0 + ((lane >> 4) << 3);
  v8h lo = *(const v8h*)(base);        // K offsets +0..7  (or +8..15)
  v8h hi = *(const v8h*)(base + 16);   // K offsets +16..23 (or +24..31)
  return __builtin_shufflevector(lo, hi, 0, 1, 2, 3, 4, 5, 6, 7, 8, 9, 10, 11,
                                 12, 13, 14, 15);
}

// K index inside a fragment for element e of this lane (matches frag_load).
DEV int frag_k(int lane, int e) {
  return ((lane >> 4) << 3) + (e & 7) + ((e >> 3) << 4);
}

// ---------------------------------------------------------------------------
// f32 -> f16 conversion
// ---------------------------------------------------------------------------
__global__ __launch_bounds__(256) void cvt_f32_f16(const float* __restrict__ s,
                                                   _Float16* __restrict__ d,
                                                   size_t n) {
  size_t i = (size_t)blockIdx.x * 256 + threadIdx.x;
  if (i < n) d[i] = (_Float16)s[i];
}

// ---------------------------------------------------------------------------
// C[M,N] = A[M,K] @ B[N,K]^T + bias[N].
// Block tile 256x64, K-tile 64, double-buffered async LDS staging.
// 8 waves, each computing a 32x64 strip (2x4 accumulator tiles, 16 WMMAs per
// K-tile).  Per block each 64-K step moves exactly 40KB once (no intra-block
// redundancy); LDS rows padded to 72 halves for conflict-free ds_load_b128.
// ---------------------------------------------------------------------------
__global__ __launch_bounds__(256) void gemm_rrt(const _Float16* __restrict__ A,
                                                const _Float16* __restrict__ B,
                                                const float* __restrict__ bias,
                                                float* __restrict__ C, int M,
                                                int N, int K) {
  __shared__ __align__(16) _Float16 sA[2][256][72];
  __shared__ __align__(16) _Float16 sB[2][64][72];
  const int t = threadIdx.x, w = t >> 5, lane = t & 31;
  const int mblk = blockIdx.x * 256;
  const int nblk = blockIdx.y * 64;

  auto stage = [&](int buf, int k0) {
    // A tile: 256 rows x 64 halves = 2048 16B chunks; 8 per thread.
#pragma unroll
    for (int q = 0; q < 8; ++q) {
      const int id = q * 256 + t;
      const int row = id >> 3, c = id & 7;
      cp16_to_lds(&sA[buf][row][c * 8],
                  A + (size_t)(mblk + row) * K + k0 + c * 8);
    }
    // B tile: 64 rows x 64 halves = 512 chunks; 2 per thread.
#pragma unroll
    for (int q = 0; q < 2; ++q) {
      const int id = q * 256 + t;
      const int row = id >> 3, c = id & 7;
      cp16_to_lds(&sB[buf][row][c * 8],
                  B + (size_t)(nblk + row) * K + k0 + c * 8);
    }
  };

  v8f acc[2][4] = {};                       // 32x64 per wave
  const int nk = K >> 6;
  stage(0, 0);
  wait_lds_cp();
  __syncthreads();
  for (int kt = 0; kt < nk; ++kt) {
    const int cur = kt & 1;
    if (kt + 1 < nk) stage(cur ^ 1, (kt + 1) << 6);   // prefetch next K-tile
#pragma unroll
    for (int k0 = 0; k0 < 64; k0 += 32) {
      v16h a0 = frag_load(&sA[cur][0][0], 72, w * 32, k0);
      v16h a1 = frag_load(&sA[cur][0][0], 72, w * 32 + 16, k0);
#pragma unroll
      for (int j = 0; j < 4; ++j) {
        v16h b = frag_load(&sB[cur][0][0], 72, j * 16, k0);
        acc[0][j] = wmma32(a0, b, acc[0][j]);
        acc[1][j] = wmma32(a1, b, acc[1][j]);
      }
    }
    wait_lds_cp();
    __syncthreads();
  }
  const int half = lane >> 4, col = lane & 15;
#pragma unroll
  for (int j = 0; j < 4; ++j) {
    const int n = nblk + j * 16 + col;
    const float bv = bias[n];
#pragma unroll
    for (int v = 0; v < 8; ++v) {
      const int m0 = mblk + w * 32 + half * 8 + v;  // D: lanes<16 M=v, else v+8
      C[(size_t)m0 * N + n] = acc[0][j][v] + bv;
      C[(size_t)(m0 + 16) * N + n] = acc[1][j][v] + bv;
    }
  }
}

// ---------------------------------------------------------------------------
// Post-projection pointwise: dt=softplus(dt+dt_bias), dA=-exp(A_log)*dt,
// X16 = f16(x*dt), B/C chunks to f16.  One block per (b,l) row.
// ---------------------------------------------------------------------------
__global__ __launch_bounds__(256) void ssd_post(const float* __restrict__ xBCdt,
                                                const float* __restrict__ A_log,
                                                const float* __restrict__ dt_bias,
                                                _Float16* __restrict__ X16,
                                                _Float16* __restrict__ B16,
                                                _Float16* __restrict__ C16,
                                                float* __restrict__ dA) {
  const int r = blockIdx.x;             // 0..8191 = b*4096 + l
  const int t = threadIdx.x;
  __shared__ float dts[64];
  const float* row = xBCdt + (size_t)r * 4416;
  if (t < 64) {
    const float z = row[4352 + t] + dt_bias[t];
    const float sp = (z > 20.0f) ? z : log1pf(__expf(z));
    dts[t] = sp;
    dA[(size_t)r * 64 + t] = -__expf(A_log[t]) * sp;
  }
  __syncthreads();
  for (int i = t; i < 4096; i += 256)
    X16[(size_t)r * 4096 + i] = (_Float16)(row[i] * dts[i >> 6]);
  if (t < 128)
    B16[(size_t)r * 128 + t] = (_Float16)row[4096 + t];
  else
    C16[(size_t)r * 128 + (t - 128)] = (_Float16)row[4224 + (t - 128)];
}

// Per-(b,chunk,head) cumulative sum of dA over the 64-step chunk.
__global__ __launch_bounds__(256) void ssd_cumsum(const float* __restrict__ dA,
                                                  float* __restrict__ Acs) {
  const int g = blockIdx.x * 256 + threadIdx.x;   // 0..8191 = bc*64 + h
  if (g >= 8192) return;
  const int h = g & 63, bc = g >> 6;
  float acc = 0.0f;
  for (int l = 0; l < 64; ++l) {
    acc += dA[(size_t)(bc * 64 + l) * 64 + h];
    Acs[((size_t)bc * 64 + h) * 64 + l] = acc;     // [bc][h][l]
  }
}

// ---------------------------------------------------------------------------
// Intra-chunk:  G = C@B^T once per (b,chunk); then per head:
//   Y_diag = (G .* exp(segsum)) @ X        (masked decay built as A-fragment)
//   states[p,n] = sum_l exp(Acs[63]-Acs[l]) * x[l,p] * B[l,n]
// grid (128 = b*chunks, 8 head-groups), 256 threads / 8 waves.
// ---------------------------------------------------------------------------
__global__ __launch_bounds__(256) void ssd_intra(const _Float16* __restrict__ C16,
                                                 const _Float16* __restrict__ B16,
                                                 const _Float16* __restrict__ X16,
                                                 const float* __restrict__ Acs,
                                                 float* __restrict__ states,
                                                 _Float16* __restrict__ Y16) {
  const int bc = blockIdx.x;            // b*64 + c
  const int hg = blockIdx.y;            // head group of 8
  const int t = threadIdx.x, w = t >> 5, lane = t & 31;
  const int rowbase = bc * 64;          // = b*4096 + c*64

  __shared__ __align__(16) _Float16 sBT[128][72];   // B^T : [n][l]
  __shared__ __align__(16) _Float16 sXT[64][72];    // x^T : [p][l] (per head)
  __shared__ __align__(16) float sG[64][65];        // C@B^T
  __shared__ float sA[64];                          // Acs for current head

  for (int i = t; i < 64 * 128; i += 256) {
    const int l = i >> 7, n = i & 127;
    sBT[n][l] = B16[(size_t)(rowbase + l) * 128 + n];
  }
  __syncthreads();

  {  // G = C @ B^T (fragments straight from global; chunk rows are contiguous)
    const int lt = w >> 1, st0 = (w & 1) * 2;
    const _Float16* Cp = C16 + (size_t)rowbase * 128;
    const _Float16* Bp = B16 + (size_t)rowbase * 128;
    v8f g0 = {}, g1 = {};
#pragma unroll
    for (int k0 = 0; k0 < 128; k0 += 32) {
      v16h a = frag_load(Cp, 128, lt * 16, k0);
      v16h b0 = frag_load(Bp, 128, st0 * 16, k0);
      v16h b1 = frag_load(Bp, 128, st0 * 16 + 16, k0);
      g0 = wmma32(a, b0, g0);
      g1 = wmma32(a, b1, g1);
    }
    const int half = lane >> 4, col = lane & 15;
#pragma unroll
    for (int v = 0; v < 8; ++v) {
      const int l = lt * 16 + half * 8 + v;
      sG[l][st0 * 16 + col] = g0[v];
      sG[l][st0 * 16 + 16 + col] = g1[v];
    }
  }
  __syncthreads();

  for (int hh = 0; hh < 8; ++hh) {
    const int h = hg * 8 + hh;
    if (hh) __syncthreads();            // protect sXT/sA reuse
    for (int i = t; i < 64 * 64; i += 256) {
      const int p = i >> 6, l = i & 63;
      sXT[p][l] = X16[(size_t)(rowbase + l) * 4096 + h * 64 + p];
    }
    if (t < 64) sA[t] = Acs[((size_t)bc * 64 + h) * 64 + t];
    __syncthreads();

    const int half = lane >> 4, col = lane & 15;

    {  // ---- Y_diag ----
      const int lt = w & 3, pg = w >> 2;          // l-tile, p-half (32 cols)
      const int l = lt * 16 + (lane & 15);
      const float Al = sA[l];
      v8f y0 = {}, y1 = {};
#pragma unroll
      for (int s0 = 0; s0 < 64; s0 += 32) {
        v16h m;
#pragma unroll
        for (int e = 0; e < 16; ++e) {            // masked decay A-fragment
          const int s = s0 + frag_k(lane, e);
          const float val = (s <= l) ? sG[l][s] * __expf(Al - sA[s]) : 0.0f;
          m[e] = (_Float16)val;
        }
        v16h x0 = frag_load(&sXT[0][0], 72, pg * 32, s0);
        v16h x1 = frag_load(&sXT[0][0], 72, pg * 32 + 16, s0);
        y0 = wmma32(m, x0, y0);
        y1 = wmma32(m, x1, y1);
      }
#pragma unroll
      for (int v = 0; v < 8; ++v) {
        const int lr = lt * 16 + half * 8 + v;
        const size_t base = (size_t)(rowbase + lr) * 4096 + h * 64 + pg * 32;
        Y16[base + col] = (_Float16)y0[v];
        Y16[base + 16 + col] = (_Float16)y1[v];
      }
    }

    {  // ---- per-chunk states ----
      const int pt = w & 3, ng = w >> 2;          // p-tile, n-half (64 cols)
      const int p = pt * 16 + (lane & 15);
      const float Aend = sA[63];
      v8f sacc[4] = {};
#pragma unroll
      for (int l0 = 0; l0 < 64; l0 += 32) {
        v16h a;
#pragma unroll
        for (int e = 0; e < 16; ++e) {            // decay folded into A operand
          const int l = l0 + frag_k(lane, e);
          a[e] = (_Float16)((float)sXT[p][l] * __expf(Aend - sA[l]));
        }
#pragma unroll
        for (int j = 0; j < 4; ++j) {
          v16h b = frag_load(&sBT[0][0], 72, ng * 64 + j * 16, l0);
          sacc[j] = wmma32(a, b, sacc[j]);
        }
      }
      const size_t sbase = ((size_t)bc * 64 + h) * 64;   // [bc][h][p][n]
#pragma unroll
      for (int j = 0; j < 4; ++j)
#pragma unroll
        for (int v = 0; v < 8; ++v) {
          const int pr = pt * 16 + half * 8 + v;
          const int n = ng * 64 + j * 16 + col;
          states[(sbase + pr) * 128 + n] = sacc[j][v];
        }
    }
  }
}

// ---------------------------------------------------------------------------
// Inter-chunk scan: S <- exp(Aend_c)*S + states_c.  One thread per (b,h,p,n).
// Emits f16 states_prev (input state of each chunk) + fp32 final_state.
// ---------------------------------------------------------------------------
__global__ __launch_bounds__(256) void ssd_scan(const float* __restrict__ states,
                                                const float* __restrict__ Acs,
                                                _Float16* __restrict__ Sprev,
                                                float* __restrict__ finalst) {
  const size_t g = (size_t)blockIdx.x * 256 + threadIdx.x;
  if (g >= (size_t)2 * 64 * 64 * 128) return;
  const int n = (int)(g & 127);
  const int p = (int)((g >> 7) & 63);
  const int h = (int)((g >> 13) & 63);
  const int b = (int)(g >> 19);
  float S = 0.0f;
  for (int c = 0; c < 64; ++c) {
    const int bc = b * 64 + c;
    const size_t idx = (((size_t)bc * 64 + h) * 64 + p) * 128 + n;
    Sprev[idx] = (_Float16)S;
    const float Aend = Acs[((size_t)bc * 64 + h) * 64 + 63];
    S = __expf(Aend) * S + states[idx];
  }
  finalst[(((size_t)b * 64 + h) * 64 + p) * 128 + n] = S;   // (b,h,p,n)
}

// ---------------------------------------------------------------------------
// Y_off = exp(Acs[l]) * (C @ states_prev^T); fused add with Y_diag (in Y16).
// One block per (b,chunk,head); 16 output tiles over 8 waves.
// ---------------------------------------------------------------------------
__global__ __launch_bounds__(256) void ssd_inter(const _Float16* __restrict__ C16,
                                                 const _Float16* __restrict__ Sprev,
                                                 const float* __restrict__ Acs,
                                                 _Float16* __restrict__ Y16) {
  const int bch = blockIdx.x;           // bc*64 + h
  const int bc = bch >> 6, h = bch & 63;
  const int t = threadIdx.x, w = t >> 5, lane = t & 31;
  const int rowbase = bc * 64;
  __shared__ float sA[64];
  if (t < 64) sA[t] = Acs[(size_t)bch * 64 + t];
  __syncthreads();

  const int lt = w >> 1, pt0 = (w & 1) * 2;
  const _Float16* Cp = C16 + (size_t)rowbase * 128;
  const _Float16* Sp = Sprev + (size_t)bch * 64 * 128;   // [p][n], n-contig
  v8f y0 = {}, y1 = {};
#pragma unroll
  for (int n0 = 0; n0 < 128; n0 += 32) {
    v16h a = frag_load(Cp, 128, lt * 16, n0);
    v16h b0 = frag_load(Sp, 128, pt0 * 16, n0);
    v16h b1 = frag_load(Sp, 128, pt0 * 16 + 16, n0);
    y0 = wmma32(a, b0, y0);
    y1 = wmma32(a, b1, y1);
  }
  const int half = lane >> 4, col = lane & 15;
#pragma unroll
  for (int v = 0; v < 8; ++v) {
    const int l = lt * 16 + half * 8 + v;
    const float e = __expf(sA[l]);
    const size_t base = (size_t)(rowbase + l) * 4096 + h * 64 + pt0 * 16;
    const float r0 = y0[v] * e + (float)Y16[base + col];
    const float r1 = y1[v] * e + (float)Y16[base + 16 + col];
    Y16[base + col] = (_Float16)r0;
    Y16[base + 16 + col] = (_Float16)r1;
  }
}

// ---------------------------------------------------------------------------
extern "C" void kernel_launch(void* const* d_in, const int* in_sizes, int n_in,
                              void* d_out, int out_size, void* d_ws,
                              size_t ws_size, hipStream_t stream) {
  const float* u = (const float*)d_in[0];        // (2,4096,2048)
  const float* W_in = (const float*)d_in[1];     // (4416,2048)
  const float* b_in = (const float*)d_in[2];     // (4416,)
  const float* W_out = (const float*)d_in[3];    // (2048,4096)
  const float* b_out = (const float*)d_in[4];    // (2048,)
  const float* A_log = (const float*)d_in[5];    // (64,)
  const float* dt_bias = (const float*)d_in[6];  // (64,)
  float* y_out = (float*)d_out;                           // 8192*2048
  float* fs_out = y_out + (size_t)8192 * 2048;            // (2,64,64,128)

  char* ws = (char*)d_ws;
  size_t off = 0;
  auto alloc = [&](size_t bytes) -> void* {
    void* p = ws + off;
    off += (bytes + 255) & ~(size_t)255;
    return p;
  };
  _Float16* u16 = (_Float16*)alloc((size_t)8192 * 2048 * 2);
  _Float16* Wi16 = (_Float16*)alloc((size_t)4416 * 2048 * 2);
  _Float16* Wo16 = (_Float16*)alloc((size_t)2048 * 4096 * 2);
  float* xBCdt = (float*)alloc((size_t)8192 * 4416 * 4);
  _Float16* X16 = (_Float16*)alloc((size_t)8192 * 4096 * 2);
  _Float16* B16 = (_Float16*)alloc((size_t)8192 * 128 * 2);
  _Float16* C16 = (_Float16*)alloc((size_t)8192 * 128 * 2);
  float* dA = (float*)alloc((size_t)8192 * 64 * 4);
  float* Acs = (float*)alloc((size_t)8192 * 64 * 4);
  float* states = (float*)alloc((size_t)2 * 64 * 64 * 64 * 128 * 4);
  _Float16* Sprev = (_Float16*)alloc((size_t)2 * 64 * 64 * 64 * 128 * 2);
  _Float16* Y16 = (_Float16*)alloc((size_t)8192 * 4096 * 2);
  (void)ws_size;  // needs ~724 MB of scratch (device HBM: 432 GiB)

  cvt_f32_f16<<<65536, 256, 0, stream>>>(u, u16, (size_t)8192 * 2048);
  cvt_f32_f16<<<35328, 256, 0, stream>>>(W_in, Wi16, (size_t)4416 * 2048);
  cvt_f32_f16<<<32768, 256, 0, stream>>>(W_out, Wo16, (size_t)2048 * 4096);

  // xBCdt = u @ W_in^T + b_in
  gemm_rrt<<<dim3(32, 69), 256, 0, stream>>>(u16, Wi16, b_in, xBCdt, 8192, 4416,
                                             2048);
  ssd_post<<<8192, 256, 0, stream>>>(xBCdt, A_log, dt_bias, X16, B16, C16, dA);
  ssd_cumsum<<<32, 256, 0, stream>>>(dA, Acs);
  ssd_intra<<<dim3(128, 8), 256, 0, stream>>>(C16, B16, X16, Acs, states, Y16);
  ssd_scan<<<4096, 256, 0, stream>>>(states, Acs, Sprev, fs_out);
  ssd_inter<<<8192, 256, 0, stream>>>(C16, Sprev, Acs, Y16);
  // y = Y @ W_out^T + b_out
  gemm_rrt<<<dim3(32, 32), 256, 0, stream>>>(Y16, Wo16, b_out, y_out, 8192,
                                             2048, 4096);
}
